// WReN_47304769798289
// MI455X (gfx1250) — matrix operand order repaired and checked by reading, not compiled, not run on hardware
//
#include <hip/hip_runtime.h>
#include <hip/hip_bf16.h>

typedef __bf16 bf16_t;
typedef __attribute__((ext_vector_type(16))) __bf16 bf16x16;
typedef __attribute__((ext_vector_type(8)))  float  f32x8;

__device__ __forceinline__ bf16_t to_bf16(float f) { return (bf16_t)f; }

// Async global->LDS copy, 16 bytes per lane, tracked by ASYNCcnt.
// VDST = LDS byte address (per lane), VADDR = 64-bit global address (GV mode).
__device__ __forceinline__ void async_cp16(unsigned int lds_off, const void* g)
{
  unsigned long long ga = (unsigned long long)g;
  asm volatile("global_load_async_to_lds_b128 %0, %1, off"
               :: "v"(lds_off), "v"(ga) : "memory");
}
// Wait until ASYNCcnt <= n (compile-time constant n).
__device__ __forceinline__ void wait_async_le(int n)
{
  if (n == 0)      asm volatile("s_wait_asynccnt 0x0" ::: "memory");
  else if (n == 2) asm volatile("s_wait_asynccnt 0x2" ::: "memory");
  else             asm volatile("s_wait_asynccnt 0x4" ::: "memory");
}

// ---------------------------------------------------------------------------
// Generic bf16 WMMA GEMM:  C[M,N] = epilogue(A[M,K] @ B[K,N] + bias)
//   A: row-major (bf16 staged via async-to-LDS, or f32 converted on the fly)
//   B: pre-transposed N-major bf16 (Bt[n*ldb + k]), zero-padded, async-to-LDS
//   OUT: 0 = f32 store, 1 = bf16 store, 2 = relu + atomic scatter-add into
//        per-batch S0 / per-answer S1 relation sums (fused layer-3 reduce)
// Block 128 x BN, K-step 32, 256 threads = 8 waves in a 4x2 grid,
// per wave 32 x (BN/2) = 2 x (BN/32) v_wmma_f32_16x16x32_bf16 tiles.
// Double-buffered LDS tiles: next tile's async DMA overlaps current WMMAs;
// every wave issues exactly NASYNC ops per stage so the s_wait_asynccnt
// immediate is wave-uniform (OOB B rows go to LDS slack, clamped source).
// ---------------------------------------------------------------------------
template<int BM, int BN, bool A_F32, int OUT, bool BIAS, bool RELU>
__global__ __launch_bounds__(256) void gemm_wmma(
    const void* __restrict__ Aptr, const bf16_t* __restrict__ Bt,
    const float* __restrict__ bias, void* __restrict__ Cptr,
    int K, int lda, int ldb, int ldc, int biasN,
    float* __restrict__ S0, float* __restrict__ S1)
{
  constexpr int SA = 40;            // LDS row stride (bf16) w/ pad (80 B)
  constexpr int SB = 40;
  constexpr int BSLOTS = 128;       // B rows allocated per buffer (slack for BN=96)
  constexpr int WN = BN / 2;
  constexpr int TN = WN / 16;
  constexpr int TM = 2;
  constexpr int NASYNC = A_F32 ? 2 : 4;   // async ops per wave per stage

  __shared__ bf16_t Asm[2 * BM * SA];
  __shared__ bf16_t Bsm[2 * BSLOTS * SB];

  const int tid  = threadIdx.x;
  const int lane = tid & 31;
  const int wave = tid >> 5;
  const int wrow = wave >> 1;       // 0..3  (M direction)
  const int wcol = wave & 1;        // 0..1  (N direction)
  const int m0 = blockIdx.y * BM;
  const int n0 = blockIdx.x * BN;

  const unsigned int AsmBase = (unsigned int)(size_t)&Asm[0];
  const unsigned int BsmBase = (unsigned int)(size_t)&Bsm[0];

  f32x8 acc[TM][TN];
  #pragma unroll
  for (int i = 0; i < TM; ++i)
    #pragma unroll
    for (int j = 0; j < TN; ++j)
      #pragma unroll
      for (int v = 0; v < 8; ++v) acc[i][j][v] = 0.f;

  const int ar = tid >> 1;          // f32-A path: A-tile row 0..127
  const int ah = (tid & 1) * 16;    // f32-A path: K half-offset

  auto stage = [&](int buf, int k0) {
    // ---- A tile (BM x 32 bf16) ----
    if (A_F32) {
      const float* Af = (const float*)Aptr + (size_t)(m0 + ar) * lda + (k0 + ah);
      bf16_t* dst = &Asm[buf * BM * SA + ar * SA + ah];
      #pragma unroll
      for (int e = 0; e < 4; ++e) {
        float4 f = ((const float4*)Af)[e];
        dst[4*e+0] = to_bf16(f.x); dst[4*e+1] = to_bf16(f.y);
        dst[4*e+2] = to_bf16(f.z); dst[4*e+3] = to_bf16(f.w);
      }
      __builtin_prefetch(Af + 32, 0, 1);
    } else {
      const bf16_t* Ab = (const bf16_t*)Aptr + (size_t)m0 * lda + k0;
      const unsigned abase = AsmBase + (unsigned)(buf * BM * SA * 2);
      #pragma unroll
      for (int i2 = 0; i2 < 2; ++i2) {
        const int lin = i2 * 256 + tid;
        const int row = lin >> 2, ch = lin & 3;
        async_cp16(abase + (unsigned)(row * (SA * 2) + ch * 16),
                   Ab + (size_t)row * lda + ch * 8);
      }
    }
    // ---- B tile (BN x 32 bf16, N-major) via async-to-LDS ----
    {
      const bf16_t* Bb = Bt + (size_t)n0 * ldb + k0;
      const unsigned bbase = BsmBase + (unsigned)(buf * BSLOTS * SB * 2);
      #pragma unroll
      for (int i2 = 0; i2 < 2; ++i2) {
        const int lin = i2 * 256 + tid;
        const int row = lin >> 2, ch = lin & 3;
        const int srow = (row < BN) ? row : (BN - 1);  // clamp; dst hits slack
        async_cp16(bbase + (unsigned)(row * (SB * 2) + ch * 16),
                   Bb + (size_t)srow * ldb + ch * 8);
      }
    }
  };

  int buf = 0;
  stage(0, 0);
  for (int k0 = 0; k0 < K; k0 += 32) {
    const bool has_next = (k0 + 32) < K;
    if (has_next) stage(buf ^ 1, k0 + 32);   // DMA next tile during compute
    if (has_next) wait_async_le(NASYNC);     // only current tile must be done
    else          wait_async_le(0);
    __syncthreads();                         // publish tile (also waits DScnt)

    const bf16_t* As = &Asm[buf * BM * SA];
    const bf16_t* Bs = &Bsm[buf * BSLOTS * SB];

    // ---- fragments from LDS per documented CDNA5 VGPR layouts ----
    const int h  = lane >> 4;
    const int rr = lane & 15;
    union { bf16x16 v; uint4 u[2]; } fa[TM], fb[TN];
    #pragma unroll
    for (int i = 0; i < TM; ++i) {
      const int row = wrow * 32 + i * 16 + rr;
      fa[i].u[0] = *(const uint4*)&As[row * SA + h * 8];        // K h*8..h*8+7
      fa[i].u[1] = *(const uint4*)&As[row * SA + 16 + h * 8];   // K 16+h*8..
    }
    #pragma unroll
    for (int j = 0; j < TN; ++j) {
      const int col = wcol * WN + j * 16 + rr;
      fb[j].u[0] = *(const uint4*)&Bs[col * SB + h * 8];        // K h*8..h*8+7
      fb[j].u[1] = *(const uint4*)&Bs[col * SB + 16 + h * 8];
    }
    #pragma unroll
    for (int i = 0; i < TM; ++i)
      #pragma unroll
      for (int j = 0; j < TN; ++j)
        acc[i][j] = __builtin_amdgcn_wmma_f32_16x16x32_bf16(
            false, fa[i].v, false, fb[j].v, (short)0, acc[i][j], false, false);

    __syncthreads();   // reads done; next stage may overwrite this buffer
    buf ^= 1;
  }

  // ---- epilogue ----
  const int h = lane >> 4;
  #pragma unroll
  for (int i = 0; i < TM; ++i) {
    const int tm = m0 + wrow * 32 + i * 16;
    #pragma unroll
    for (int j = 0; j < TN; ++j) {
      const int tn   = n0 + wcol * WN + j * 16;
      const int colx = tn + (lane & 15);
      float bv = 0.f;
      if (BIAS) bv = (colx < biasN) ? bias[colx] : 0.f;
      #pragma unroll
      for (int v = 0; v < 8; ++v) {
        float x = acc[i][j][v] + bv;
        if (RELU) x = fmaxf(x, 0.f);
        const int row = tm + v + h * 8;
        if (OUT == 0) {
          ((float*)Cptr)[(size_t)row * ldc + colx] = x;
        } else if (OUT == 1) {
          ((bf16_t*)Cptr)[(size_t)row * ldc + colx] = to_bf16(x);
        } else {
          const int q = row & 127, bb = row >> 7;   // rows are b*128 + pair
          float* dst = (q < 64)
              ? &S0[(size_t)bb * ldc + colx]                          // ctx_g
              : &S1[(size_t)(bb * 8 + ((q - 64) >> 3)) * ldc + colx]; // cc_g[a]
          atomicAdd(dst, x);
        }
      }
    }
  }
}

// f32 weight (K x N) -> bf16 transposed + padded (Np x Kp), N-major
__global__ void convert_wT(const float* __restrict__ src, bf16_t* __restrict__ dst,
                           int K, int N, int Kp, int Np)
{
  size_t idx = (size_t)blockIdx.x * blockDim.x + threadIdx.x;
  size_t total = (size_t)Np * Kp;
  if (idx >= total) return;
  int k = (int)(idx % Kp), n = (int)(idx / Kp);
  float v = (k < K && n < N) ? src[(size_t)k * N + n] : 0.f;
  dst[idx] = to_bf16(v);
}

// tagged[8192 x 544]: cols 512..543 = one-hot tag (eye9[p<8?p:8]) then zero pad
__global__ void init_tags(bf16_t* __restrict__ tagged)
{
  int idx = blockIdx.x * blockDim.x + threadIdx.x;
  if (idx >= 8192 * 32) return;
  int c = idx & 31, r = idx >> 5, p = r & 15;
  float v = 0.f;
  if (c < 9) { int t = (p < 8) ? p : 8; v = (c == t) ? 1.f : 0.f; }
  tagged[(size_t)r * 544 + 512 + c] = to_bf16(v);
}

// H1[b,q,c] = relu(P[b,j] + Q[b,i] + bg1)  (layer-1 via concat decomposition)
__global__ void combine_h1(const float* __restrict__ P, const float* __restrict__ Q,
                           const float* __restrict__ bg1, bf16_t* __restrict__ H1)
{
  size_t idx = ((size_t)blockIdx.x * blockDim.x + threadIdx.x) * 4;
  if (idx >= (size_t)512 * 128 * 1056) return;
  int c = (int)(idx % 1056);
  size_t r = idx / 1056;
  int q = (int)(r & 127), b = (int)(r >> 7);
  int j = (q < 64) ? (q & 7) : ((q - 64) & 7);         // first half: ctx_j
  int i = (q < 64) ? (q >> 3) : (8 + ((q - 64) >> 3)); // second half: ctx_i / cho_a
  const float4 p  = *(const float4*)&P[(size_t)(b * 16 + j) * 1056 + c];
  const float4 qq = *(const float4*)&Q[(size_t)(b * 16 + i) * 1056 + c];
  float4 bb;
  bb.x = (c + 0 < 1042) ? bg1[c + 0] : 0.f;
  bb.y = (c + 1 < 1042) ? bg1[c + 1] : 0.f;
  bb.z = (c + 2 < 1042) ? bg1[c + 2] : 0.f;
  bb.w = (c + 3 < 1042) ? bg1[c + 3] : 0.f;
  bf16_t* o = &H1[idx];
  o[0] = to_bf16(fmaxf(p.x + qq.x + bb.x, 0.f));
  o[1] = to_bf16(fmaxf(p.y + qq.y + bb.y, 0.f));
  o[2] = to_bf16(fmaxf(p.z + qq.z + bb.z, 0.f));
  o[3] = to_bf16(fmaxf(p.w + qq.w + bb.w, 0.f));
}

// relations (bf16) = S0[b] + S1[b,a]
__global__ void relb_ker(const float* __restrict__ S0, const float* __restrict__ S1,
                         bf16_t* __restrict__ Rb)
{
  size_t idx = (size_t)blockIdx.x * blockDim.x + threadIdx.x;
  if (idx >= (size_t)4096 * 1056) return;
  int c = (int)(idx % 1056);
  int r = (int)(idx / 1056);
  int b = r >> 3;
  Rb[idx] = to_bf16(S0[(size_t)b * 1056 + c] + S1[idx]);
}

__global__ void zero_ker(float* __restrict__ p, size_t n)
{
  size_t i = (size_t)blockIdx.x * blockDim.x + threadIdx.x;
  if (i < n) p[i] = 0.f;
}

// f_out[r] = h[r,:1042] . Wf2 + bf2   (one wave per row, shuffle reduction)
__global__ __launch_bounds__(256) void fout_ker(const float* __restrict__ h,
                                                const float* __restrict__ Wf2,
                                                const float* __restrict__ bf2,
                                                float* __restrict__ fout)
{
  int wv = (int)((blockIdx.x * blockDim.x + threadIdx.x) >> 5);
  int lane = threadIdx.x & 31;
  if (wv >= 4096) return;
  const float* row = h + (size_t)wv * 1056;
  float s = 0.f;
  for (int c = lane; c < 1042; c += 32) s += row[c] * Wf2[c];
  #pragma unroll
  for (int off = 16; off > 0; off >>= 1) s += __shfl_down(s, off, 32);
  if (lane == 0) fout[wv] = s + bf2[0];
}

// BCE-with-logits loss sum + accuracy count -> partial[0], partial[1]
__global__ void loss_ker(const float* __restrict__ fout, const int* __restrict__ labels,
                         float* __restrict__ partial)
{
  int b = blockIdx.x * blockDim.x + threadIdx.x;
  if (b >= 512) return;
  const float* f = fout + (size_t)b * 8;
  int lab = labels[b];
  float ls = 0.f, best = f[0];
  int arg = 0;
  #pragma unroll
  for (int a = 0; a < 8; ++a) {
    float v = f[a];
    ls += fmaxf(v, 0.f) - v * ((a == lab) ? 1.f : 0.f) + log1pf(expf(-fabsf(v)));
    if (v > best) { best = v; arg = a; }
  }
  atomicAdd(&partial[0], ls);
  atomicAdd(&partial[1], (arg == lab) ? 1.f : 0.f);
}

__global__ void fin_ker(const float* __restrict__ partial, float* __restrict__ out)
{
  out[0] = partial[0] / 4096.f;  // mean over (B=512, N_ANS=8)
  out[1] = partial[1] / 512.f;   // mean over B
}

// ---------------------------------------------------------------------------
extern "C" void kernel_launch(void* const* d_in, const int* in_sizes, int n_in,
                              void* d_out, int out_size, void* d_ws, size_t ws_size,
                              hipStream_t stream)
{
  (void)in_sizes; (void)n_in; (void)out_size; (void)ws_size;
  const float* x     = (const float*)d_in[0];
  const float* W_emb = (const float*)d_in[1];
  const float* b_emb = (const float*)d_in[2];
  const float* Wg1   = (const float*)d_in[3];
  const float* bg1   = (const float*)d_in[4];
  const float* Wg2   = (const float*)d_in[5];
  const float* bg2   = (const float*)d_in[6];
  const float* Wg3   = (const float*)d_in[7];
  const float* bg3   = (const float*)d_in[8];
  const float* Wf1   = (const float*)d_in[9];
  const float* bf1   = (const float*)d_in[10];
  const float* Wf2   = (const float*)d_in[11];
  const float* bf2   = (const float*)d_in[12];
  const int*   lbl   = (const int*)d_in[13];
  float* out = (float*)d_out;

  // workspace carve-up (256B aligned)
  char* ws = (char*)d_ws;
  size_t off = 0;
  auto alloc = [&](size_t bytes) { char* p = ws + off; off = (off + bytes + 255) & ~(size_t)255; return p; };
  bf16_t* Wembt  = (bf16_t*)alloc((size_t)512  * 6400 * 2);
  bf16_t* Wg1t   = (bf16_t*)alloc((size_t)1056 * 544  * 2);
  bf16_t* Wg1b   = (bf16_t*)alloc((size_t)1056 * 544  * 2);
  bf16_t* Wg2t   = (bf16_t*)alloc((size_t)1056 * 1056 * 2);
  bf16_t* Wg3t   = (bf16_t*)alloc((size_t)1056 * 1056 * 2);
  bf16_t* Wf1t   = (bf16_t*)alloc((size_t)1056 * 1056 * 2);
  bf16_t* tagged = (bf16_t*)alloc((size_t)8192 * 544  * 2);
  float*  Pf     = (float*) alloc((size_t)8192 * 1056 * 4);
  float*  Qf     = (float*) alloc((size_t)8192 * 1056 * 4);
  bf16_t* H1     = (bf16_t*)alloc((size_t)65536 * 1056 * 2);
  bf16_t* H2     = (bf16_t*)alloc((size_t)65536 * 1056 * 2);
  float*  S0     = (float*) alloc((size_t)512  * 1056 * 4);
  float*  S1     = (float*) alloc((size_t)4096 * 1056 * 4);
  bf16_t* Rb     = (bf16_t*)alloc((size_t)4096 * 1056 * 2);
  float*  hbuf   = (float*) alloc((size_t)4096 * 1056 * 4);
  float*  fout   = (float*) alloc((size_t)4096 * 4);
  float*  part   = (float*) alloc(2 * 4);

  auto blocks = [](size_t n) { return (unsigned)((n + 255) / 256); };

  // weight conversion (bf16, transposed N-major, zero-padded)
  convert_wT<<<blocks((size_t)512 * 6400), 256, 0, stream>>>(W_emb, Wembt, 6400, 512, 6400, 512);
  convert_wT<<<blocks((size_t)1056 * 544), 256, 0, stream>>>(Wg1,              Wg1t, 521, 1042, 544, 1056);
  convert_wT<<<blocks((size_t)1056 * 544), 256, 0, stream>>>(Wg1 + 521 * 1042, Wg1b, 521, 1042, 544, 1056);
  convert_wT<<<blocks((size_t)1056 * 1056), 256, 0, stream>>>(Wg2, Wg2t, 1042, 1042, 1056, 1056);
  convert_wT<<<blocks((size_t)1056 * 1056), 256, 0, stream>>>(Wg3, Wg3t, 1042, 1042, 1056, 1056);
  convert_wT<<<blocks((size_t)1056 * 1056), 256, 0, stream>>>(Wf1, Wf1t, 1042, 1042, 1056, 1056);
  init_tags<<<blocks(8192 * 32), 256, 0, stream>>>(tagged);

  // embedding: tagged[:, :512] = x @ W_emb + b_emb   (A read as f32, cast on the fly)
  gemm_wmma<128, 128, true, 1, true, false><<<dim3(4, 64), 256, 0, stream>>>(
      x, Wembt, b_emb, tagged, 6400, 6400, 6400, 544, 512, nullptr, nullptr);

  // layer-1 halves: P = tagged @ Wg1[:521], Q = tagged @ Wg1[521:]
  gemm_wmma<128, 96, false, 0, false, false><<<dim3(11, 64), 256, 0, stream>>>(
      tagged, Wg1t, nullptr, Pf, 544, 544, 544, 1056, 0, nullptr, nullptr);
  gemm_wmma<128, 96, false, 0, false, false><<<dim3(11, 64), 256, 0, stream>>>(
      tagged, Wg1b, nullptr, Qf, 544, 544, 544, 1056, 0, nullptr, nullptr);

  // H1 = relu(P_j + Q_i + bg1) over 128 pairs per batch
  combine_h1<<<blocks((size_t)512 * 128 * 1056 / 4), 256, 0, stream>>>(Pf, Qf, bg1, H1);

  // zero relation accumulators
  zero_ker<<<blocks((size_t)512 * 1056), 256, 0, stream>>>(S0, (size_t)512 * 1056);
  zero_ker<<<blocks((size_t)4096 * 1056), 256, 0, stream>>>(S1, (size_t)4096 * 1056);
  zero_ker<<<1, 256, 0, stream>>>(part, 2);

  // g-MLP layers 2 and 3 (layer 3 fused with relation reduction)
  gemm_wmma<128, 96, false, 1, true, true><<<dim3(11, 512), 256, 0, stream>>>(
      H1, Wg2t, bg2, H2, 1056, 1056, 1056, 1056, 1042, nullptr, nullptr);
  gemm_wmma<128, 96, false, 2, true, true><<<dim3(11, 512), 256, 0, stream>>>(
      H2, Wg3t, bg3, nullptr, 1056, 1056, 1056, 1056, 1042, S0, S1);

  // relations -> f-MLP
  relb_ker<<<blocks((size_t)4096 * 1056), 256, 0, stream>>>(S0, S1, Rb);
  gemm_wmma<128, 96, false, 0, true, true><<<dim3(11, 32), 256, 0, stream>>>(
      Rb, Wf1t, bf1, hbuf, 1056, 1056, 1056, 1056, 1042, nullptr, nullptr);

  fout_ker<<<512, 256, 0, stream>>>(hbuf, Wf2, bf2, fout);
  loss_ker<<<2, 256, 0, stream>>>(fout, lbl, part);
  fin_ker<<<1, 1, 0, stream>>>(part, out);
}
